// EnhancedDTNN_29274497089904
// MI455X (gfx1250) — compile-verified, matrix operand order checked by reading, not run on hardware
//
#include <hip/hip_runtime.h>
#include <hip/hip_bf16.h>
#include <cmath>

// ---------------------------------------------------------------------------
// EnhancedDTNN fused MPNN for gfx1250 (MI455X).
//  - f16 WMMA (f32 accumulate) for all edge GEMMs
//  - weights pre-converted to transposed f16 in workspace
//  - per-layer fused kernel: gather -> 5 GEMMs -> tanh -> f32 atomic scatter
//  - weight staging via GLOBAL_LOAD_ASYNC_TO_LDS_B128 (ASYNCcnt)
//  - depth-2 software pipeline on B fragments to avoid dscnt==0 stalls
// ---------------------------------------------------------------------------

typedef _Float16 half_t;
typedef __attribute__((ext_vector_type(16))) _Float16 v16h;
typedef __attribute__((ext_vector_type(8)))  _Float16 v8h;
typedef __attribute__((ext_vector_type(4)))  _Float16 v4h;
typedef __attribute__((ext_vector_type(8)))  float    v8f;

#define DIMC 128
#define NCENT 30
#define LDA 160                   // K-stride of activation staging (covers 158-pad)
#define WAVES 8
#define EPB (WAVES * 16)          // edges per block
#define PER_LAYER_HALFS 86016     // 4*128*128 + 128*160
#define OFF_WN1 0
#define OFF_WN2 16384
#define OFF_WE1 32768             // 128 rows x 160 K (padded from 158)
#define OFF_WE2 53248
#define OFF_WC  69632
#define SMEM_BYTES ((20480 + 2 * WAVES * 16 * LDA) * 2)   // 122880 B

// ---- fragment load: row-major [row][k] f16 tile in LDS --------------------
// A layout (16x32 f16): lanes 0-15 = rows 0-15 holding K [k0..k0+7],[k0+16..k0+23];
// lanes 16-31 = same rows holding K [k0+8..k0+15],[k0+24..k0+31].
// B uses identical addressing with "row" = output channel n (weights stored W^T).
__device__ __forceinline__ v16h frag_load(const half_t* base, int ld, int row, int hi, int k0) {
  const half_t* p = base + row * ld + k0 + hi * 8;
  v8h lo = *(const v8h*)(p);
  v8h hh = *(const v8h*)(p + 16);
  return __builtin_shufflevector(lo, hh, 0,1,2,3,4,5,6,7,8,9,10,11,12,13,14,15);
}

// one wave: [16 x K] @ [K x 128] -> acc[8] (8 N-tiles of 16x16 f32)
// B fragments pipelined at depth 2 so ds_load latency hides under WMMA issue.
__device__ __forceinline__ void wave_gemm(const half_t* A, int lda,
                                          const half_t* W, int ldw,
                                          int nk, int r16, int hi, v8f acc[8]) {
  #pragma unroll
  for (int kt = 0; kt < nk; ++kt) {
    const int k0 = kt * 32;
    v16h a  = frag_load(A, lda, r16, hi, k0);
    v16h b0 = frag_load(W, ldw, 0 * 16 + r16, hi, k0);
    v16h b1 = frag_load(W, ldw, 1 * 16 + r16, hi, k0);
    #pragma unroll
    for (int nt = 0; nt < 8; ++nt) {
      v16h bn = b1;
      if (nt < 6) bn = frag_load(W, ldw, (nt + 2) * 16 + r16, hi, k0);
      acc[nt] = __builtin_amdgcn_wmma_f32_16x16x32_f16(
          false, a, false, b0, (short)0, acc[nt], false, false);
      b0 = b1; b1 = bn;
    }
  }
}

// ---- async weight staging: global -> LDS direct (no VGPR round trip) ------
// s_wgt lives at wave-relative LDS byte offset 0 (only dynamic LDS in kernel).
// GVS addressing: mem = SGPR_base + VGPR_offset; LDS dest = VGPR value.
__device__ __forceinline__ void stage_weights_async(uint32_t lds_byte_off,
                                                    const half_t* src,
                                                    int nhalfs, int tid) {
  for (int i = tid * 8; i < nhalfs; i += 256 * 8) {
    uint32_t ldsa = lds_byte_off + (uint32_t)i * 2u;
    uint32_t voff = (uint32_t)i * 2u;
    asm volatile("global_load_async_to_lds_b128 %0, %1, %2"
                 :: "v"(ldsa), "v"(voff), "s"(src)
                 : "memory");
  }
  asm volatile("s_wait_asynccnt 0x0" ::: "memory");
}

// ---------------------------------------------------------------------------
__global__ void embed_kernel(const int* __restrict__ node_types,
                             const float* __restrict__ node_emb,
                             float* __restrict__ h, int total) {
  int idx = blockIdx.x * blockDim.x + threadIdx.x;
  if (idx >= total) return;
  int node = idx >> 7, c = idx & 127;
  h[idx] = node_emb[(size_t)node_types[node] * DIMC + c];
}

__global__ void convert_weights_kernel(const float* __restrict__ Wn1,
                                       const float* __restrict__ Wn2,
                                       const float* __restrict__ We1,
                                       const float* __restrict__ We2,
                                       const float* __restrict__ Wc,
                                       half_t* __restrict__ wf) {
  int idx = blockIdx.x * blockDim.x + threadIdx.x;
  if (idx >= 3 * PER_LAYER_HALFS) return;
  int l = idx / PER_LAYER_HALFS;
  int o = idx - l * PER_LAYER_HALFS;
  float v;
  if (o < OFF_WN2) {                    // Wn1^T
    int q = o - OFF_WN1, n = q / 128, k = q % 128;
    v = Wn1[(size_t)l * 16384 + k * 128 + n];
  } else if (o < OFF_WE1) {             // Wn2^T
    int q = o - OFF_WN2, n = q / 128, k = q % 128;
    v = Wn2[(size_t)l * 16384 + k * 128 + n];
  } else if (o < OFF_WE2) {             // We1^T, K padded 158 -> 160
    int q = o - OFF_WE1, n = q / 160, k = q % 160;
    v = (k < 158) ? We1[(size_t)l * 158 * 128 + k * 128 + n] : 0.0f;
  } else if (o < OFF_WC) {              // We2^T
    int q = o - OFF_WE2, n = q / 128, k = q % 128;
    v = We2[(size_t)l * 16384 + k * 128 + n];
  } else {                              // Wc^T
    int q = o - OFF_WC, n = q / 128, k = q % 128;
    v = Wc[(size_t)l * 16384 + k * 128 + n];
  }
  wf[idx] = (half_t)v;
}

__global__ void zero_kernel(float* __restrict__ p, int n) {
  int i = blockIdx.x * blockDim.x + threadIdx.x;
  if (i < n) p[i] = 0.0f;
}

// ---------------------------------------------------------------------------
__global__ void __launch_bounds__(256) conv_kernel(
    const float* __restrict__ h_in, float* __restrict__ h_out,
    const int* __restrict__ srcIdx, const int* __restrict__ dstIdx,
    const int* __restrict__ edge_types, const float* __restrict__ distances,
    const float* __restrict__ edge_emb,
    const half_t* __restrict__ wgt,     // this layer's transposed f16 weights
    const float* __restrict__ bn1, const float* __restrict__ bn2,
    const float* __restrict__ be1, const float* __restrict__ be2,
    const float* __restrict__ bc,
    int n_edges)
{
  extern __shared__ char smem_raw[];
  half_t* s_wgt = (half_t*)smem_raw;                                  // 20480 halfs @ LDS off 0
  half_t* s_act = (half_t*)(smem_raw + 20480 * sizeof(half_t));       // 2 bufs/wave

  const int tid  = threadIdx.x;
  const int wave = tid >> 5;
  const int lane = tid & 31;
  const int r16  = lane & 15;
  const int hi   = lane >> 4;
  half_t* buf0 = s_act + (2 * wave + 0) * (16 * LDA);
  half_t* buf1 = s_act + (2 * wave + 1) * (16 * LDA);
  const int e0 = blockIdx.x * EPB + wave * 16;

  const v8f vzero = {0.f,0.f,0.f,0.f,0.f,0.f,0.f,0.f};
  v8f acc[8];

  // ---- gather h[src] tile (f32 -> f16) into buf1 --------------------------
  #pragma unroll 4
  for (int r = 0; r < 16; ++r) {
    int e = e0 + r; if (e >= n_edges) e = n_edges - 1;
    int s = srcIdx[e];
    const float4 v = *(const float4*)(h_in + (size_t)s * DIMC + lane * 4);
    v4h p; p[0] = (half_t)v.x; p[1] = (half_t)v.y; p[2] = (half_t)v.z; p[3] = (half_t)v.w;
    *(v4h*)(buf1 + r * LDA + lane * 4) = p;
  }
  stage_weights_async(0u, wgt + OFF_WN1, 16384, tid);
  __syncthreads();

  // ---- G1: t1 = relu(hsrc @ Wn1 + bn1) -> buf0 ----------------------------
  #pragma unroll
  for (int i = 0; i < 8; ++i) acc[i] = vzero;
  wave_gemm(buf1, LDA, s_wgt, 128, 4, r16, hi, acc);
  #pragma unroll
  for (int nt = 0; nt < 8; ++nt) {
    float bv = bn1[nt * 16 + r16];
    #pragma unroll
    for (int r = 0; r < 8; ++r)
      buf0[(r + 8 * hi) * LDA + nt * 16 + r16] = (half_t)fmaxf(acc[nt][r] + bv, 0.f);
  }
  __syncthreads();
  stage_weights_async(0u, wgt + OFF_WN2, 16384, tid);
  __syncthreads();

  // ---- G2: np = t1 @ Wn2 + bn2 (kept in registers) ------------------------
  #pragma unroll
  for (int i = 0; i < 8; ++i) acc[i] = vzero;
  wave_gemm(buf0, LDA, s_wgt, 128, 4, r16, hi, acc);
  float npv[8][8];
  #pragma unroll
  for (int nt = 0; nt < 8; ++nt) {
    float bv = bn2[nt * 16 + r16];
    #pragma unroll
    for (int r = 0; r < 8; ++r) npv[nt][r] = acc[nt][r] + bv;
  }

  // ---- build edge features [emb(128) | rbf(30) | pad(2)] into buf1 --------
  #pragma unroll 4
  for (int r = 0; r < 16; ++r) {
    int e = e0 + r; if (e >= n_edges) e = n_edges - 1;
    int t = edge_types[e];
    const float4 v = *(const float4*)(edge_emb + (size_t)t * DIMC + lane * 4);
    v4h p; p[0] = (half_t)v.x; p[1] = (half_t)v.y; p[2] = (half_t)v.z; p[3] = (half_t)v.w;
    *(v4h*)(buf1 + r * LDA + lane * 4) = p;
  }
  const float gap = 10.0f / 29.0f;
  #pragma unroll 4
  for (int r = 0; r < 16; ++r) {
    int e = e0 + r; if (e >= n_edges) e = n_edges - 1;
    float d = distances[e];
    float val = 0.f;
    if (lane < NCENT) { float t = d - (float)lane * gap; val = expf(-t * t / gap); }
    buf1[r * LDA + 128 + lane] = (half_t)val;
  }
  __syncthreads();
  stage_weights_async(0u, wgt + OFF_WE1, 20480, tid);
  __syncthreads();

  // ---- G3: t2 = relu(e @ We1 + be1) -> buf0 (K = 160) ---------------------
  #pragma unroll
  for (int i = 0; i < 8; ++i) acc[i] = vzero;
  wave_gemm(buf1, LDA, s_wgt, 160, 5, r16, hi, acc);
  #pragma unroll
  for (int nt = 0; nt < 8; ++nt) {
    float bv = be1[nt * 16 + r16];
    #pragma unroll
    for (int r = 0; r < 8; ++r)
      buf0[(r + 8 * hi) * LDA + nt * 16 + r16] = (half_t)fmaxf(acc[nt][r] + bv, 0.f);
  }
  __syncthreads();
  stage_weights_async(0u, wgt + OFF_WE2, 16384, tid);
  __syncthreads();

  // ---- G4: prod = (t2 @ We2 + be2) * np -> buf1 ---------------------------
  #pragma unroll
  for (int i = 0; i < 8; ++i) acc[i] = vzero;
  wave_gemm(buf0, LDA, s_wgt, 128, 4, r16, hi, acc);
  #pragma unroll
  for (int nt = 0; nt < 8; ++nt) {
    float bv = be2[nt * 16 + r16];
    #pragma unroll
    for (int r = 0; r < 8; ++r)
      buf1[(r + 8 * hi) * LDA + nt * 16 + r16] = (half_t)((acc[nt][r] + bv) * npv[nt][r]);
  }
  __syncthreads();
  stage_weights_async(0u, wgt + OFF_WC, 16384, tid);
  __syncthreads();

  // ---- G5: m = tanh(prod @ Wc + bc); scatter-add to h_out[dst] ------------
  #pragma unroll
  for (int i = 0; i < 8; ++i) acc[i] = vzero;
  wave_gemm(buf1, LDA, s_wgt, 128, 4, r16, hi, acc);
  #pragma unroll
  for (int r = 0; r < 8; ++r) {
    int e = e0 + r + 8 * hi;
    bool valid = e < n_edges;
    int dn = dstIdx[valid ? e : 0];
    float* orow = h_out + (size_t)dn * DIMC;
    #pragma unroll
    for (int nt = 0; nt < 8; ++nt) {
      float v = tanhf(acc[nt][r] + bc[nt * 16 + r16]);
      if (valid) atomicAdd(orow + nt * 16 + r16, v);
    }
  }
}

// ---------------------------------------------------------------------------
__global__ void __launch_bounds__(128) readout_kernel(
    const float* __restrict__ h, const int* __restrict__ graph_ids,
    const float* __restrict__ Wr1, const float* __restrict__ br1,
    const float* __restrict__ Wr2, const float* __restrict__ br2,
    float* __restrict__ out, int n_nodes)
{
  __shared__ float hrow[128];
  __shared__ float red[128];
  int node = blockIdx.x;
  int tid = threadIdx.x;
  hrow[tid] = h[(size_t)node * DIMC + tid];
  __syncthreads();
  float acc = br1[tid];
  #pragma unroll 8
  for (int k = 0; k < 128; ++k) acc += hrow[k] * Wr1[k * 128 + tid];
  red[tid] = fmaxf(acc, 0.f) * Wr2[tid];
  for (int s = 64; s > 0; s >>= 1) {
    __syncthreads();
    if (tid < s) red[tid] += red[tid + s];
  }
  __syncthreads();
  if (tid == 0) atomicAdd(&out[graph_ids[node]], red[0] + br2[0]);
}

// ---------------------------------------------------------------------------
extern "C" void kernel_launch(void* const* d_in, const int* in_sizes, int n_in,
                              void* d_out, int out_size, void* d_ws, size_t ws_size,
                              hipStream_t stream) {
  const int*   node_types = (const int*)  d_in[0];
  const int*   edge_types = (const int*)  d_in[1];
  const int*   src        = (const int*)  d_in[2];
  const int*   dst        = (const int*)  d_in[3];
  const int*   graph_ids  = (const int*)  d_in[4];
  const float* distances  = (const float*)d_in[5];
  const float* node_emb   = (const float*)d_in[7];
  const float* edge_emb   = (const float*)d_in[8];
  const float* Wn1 = (const float*)d_in[9];  const float* bn1 = (const float*)d_in[10];
  const float* Wn2 = (const float*)d_in[11]; const float* bn2 = (const float*)d_in[12];
  const float* We1 = (const float*)d_in[13]; const float* be1 = (const float*)d_in[14];
  const float* We2 = (const float*)d_in[15]; const float* be2 = (const float*)d_in[16];
  const float* Wc  = (const float*)d_in[17]; const float* bc  = (const float*)d_in[18];
  const float* Wr1 = (const float*)d_in[19]; const float* br1 = (const float*)d_in[20];
  const float* Wr2 = (const float*)d_in[21]; const float* br2 = (const float*)d_in[22];

  const int N = in_sizes[0];
  const int E = in_sizes[2];
  float* out = (float*)d_out;

  char* ws = (char*)d_ws;
  float* hA = (float*)ws;
  float* hB = hA + (size_t)N * DIMC;
  half_t* wf = (half_t*)(ws + 2 * (size_t)N * DIMC * sizeof(float));
  const size_t hbytes = (size_t)N * DIMC * sizeof(float);

  (void)hipFuncSetAttribute(reinterpret_cast<const void*>(conv_kernel),
                            hipFuncAttributeMaxDynamicSharedMemorySize,
                            (int)SMEM_BYTES);

  embed_kernel<<<(N * DIMC + 255) / 256, 256, 0, stream>>>(node_types, node_emb, hA, N * DIMC);
  convert_weights_kernel<<<(3 * PER_LAYER_HALFS + 255) / 256, 256, 0, stream>>>(
      Wn1, Wn2, We1, We2, Wc, wf);
  zero_kernel<<<(out_size + 127) / 128, 128, 0, stream>>>(out, out_size);

  const int cgrid = (E + EPB - 1) / EPB;

  // layer 0: read hA, accumulate into hB (= copy of hA)
  hipMemcpyAsync(hB, hA, hbytes, hipMemcpyDeviceToDevice, stream);
  conv_kernel<<<cgrid, 256, SMEM_BYTES, stream>>>(
      hA, hB, src, dst, edge_types, distances, edge_emb,
      wf + 0 * PER_LAYER_HALFS, bn1 + 0, bn2 + 0, be1 + 0, be2 + 0, bc + 0, E);
  // layer 1: read hB, accumulate into hA (= copy of hB)
  hipMemcpyAsync(hA, hB, hbytes, hipMemcpyDeviceToDevice, stream);
  conv_kernel<<<cgrid, 256, SMEM_BYTES, stream>>>(
      hB, hA, src, dst, edge_types, distances, edge_emb,
      wf + 1 * PER_LAYER_HALFS, bn1 + 128, bn2 + 128, be1 + 128, be2 + 128, bc + 128, E);
  // layer 2: read hA, accumulate into hB (= copy of hA)
  hipMemcpyAsync(hB, hA, hbytes, hipMemcpyDeviceToDevice, stream);
  conv_kernel<<<cgrid, 256, SMEM_BYTES, stream>>>(
      hA, hB, src, dst, edge_types, distances, edge_emb,
      wf + 2 * PER_LAYER_HALFS, bn1 + 256, bn2 + 256, be1 + 256, be2 + 256, bc + 256, E);

  readout_kernel<<<N, 128, 0, stream>>>(hB, graph_ids, Wr1, br1, Wr2, br2, out, N);
}
